// AtlasOmegaBlock_31662498906975
// MI455X (gfx1250) — matrix operand order repaired and checked by reading, not compiled, not run on hardware
//
#include <hip/hip_runtime.h>
#include <hip/hip_bf16.h>

// ---------------------------------------------------------------------------
// AtlasOmega block for gfx1250 (MI455X). bf16 WMMA (f32 accum) everywhere.
// B=2, S=2048, D=512, H=4, dh=128, DF=2048, WIN=512, CW=16, P=1024.
// Scan: M/St state in VGPRs (C-fragment layout), bf16 shadow of M in LDS.
// pw window: double-buffered TDM tensor_load_to_lds, prefetched one step
// ahead so the DMA overlaps compute (wait tensorcnt<=1 retires the older
// in-order transfer).  pq_t: double-buffered GLOBAL_LOAD_ASYNC_TO_LDS_B128.
// ---------------------------------------------------------------------------

typedef __attribute__((ext_vector_type(16))) __bf16 v16bf;
typedef __attribute__((ext_vector_type(8)))  __bf16 v8bf;
typedef __attribute__((ext_vector_type(8)))  float  v8f;
typedef __attribute__((ext_vector_type(4)))  unsigned int v4u;
typedef __attribute__((ext_vector_type(8)))  int v8i;
typedef __attribute__((ext_vector_type(4)))  int v4i;
typedef __bf16 bf16_t;

union V16 {
  v16bf v;
  v8bf  h[2];
  __bf16 e[16];
};
union V8 {
  v8bf v;
  __bf16 e[8];
};

#define B_   2
#define S_   2048
#define D_   512
#define P_   1024
#define CW_  16
#define WIN_ 512
#define DF_  2048

#if __has_builtin(__builtin_amdgcn_tensor_load_to_lds)
#define HAS_TDM_ 1
#else
#define HAS_TDM_ 0
#endif

__device__ inline v8f wmma_bf(v16bf a, v16bf b, v8f c) {
  return __builtin_amdgcn_wmma_f32_16x16x32_bf16(false, a, false, b, (short)0, c,
                                                 false, false);
}

// ---------------------------------------------------------------------------
// Weight transpose + bf16 convert: Wt[n*K + k] = W[k*N + n]
// ---------------------------------------------------------------------------
__global__ __launch_bounds__(256) void tw_kernel(const float* __restrict__ W,
                                                 bf16_t* __restrict__ Wt,
                                                 int K, int N) {
  const int i = blockIdx.x * 256 + threadIdx.x;
  if (i >= K * N) return;
  const int k = i / N, n = i % N;
  Wt[(long)n * K + k] = (__bf16)W[i];
}

// ---------------------------------------------------------------------------
// LayerNorm over D=512, one 256-thread block per row; bf16 output.
// ---------------------------------------------------------------------------
__global__ __launch_bounds__(256) void ln512(const float* __restrict__ x,
                                             const float* __restrict__ g,
                                             const float* __restrict__ b,
                                             bf16_t* __restrict__ o) {
  __shared__ float red[256];
  __shared__ float stat[2];
  const int row = blockIdx.x, t = threadIdx.x;
  const float* xr = x + (long)row * D_;
  float a0 = xr[t], a1 = xr[t + 256];
  red[t] = a0 + a1;
  __syncthreads();
  for (int s = 128; s > 0; s >>= 1) {
    if (t < s) red[t] += red[t + s];
    __syncthreads();
  }
  if (t == 0) stat[0] = red[0] * (1.0f / 512.0f);
  __syncthreads();
  const float mean = stat[0];
  const float d0 = a0 - mean, d1 = a1 - mean;
  red[t] = d0 * d0 + d1 * d1;
  __syncthreads();
  for (int s = 128; s > 0; s >>= 1) {
    if (t < s) red[t] += red[t + s];
    __syncthreads();
  }
  if (t == 0) stat[1] = rsqrtf(red[0] * (1.0f / 512.0f) + 1e-5f);
  __syncthreads();
  const float rs = stat[1];
  bf16_t* orow = o + (long)row * D_;
  orow[t]       = (__bf16)(d0 * rs * g[t]       + b[t]);
  orow[t + 256] = (__bf16)(d1 * rs * g[t + 256] + b[t + 256]);
}

// ---------------------------------------------------------------------------
// WMMA GEMM: C[M,N] = act(A[M,K](bf16) @ Wt[N,K](bf16,transposed) + bias)(+res)
// ACT: 0 none, 1 gelu(tanh), 2 sigmoid.
// OB:  0 -> f32 row-major, 1 -> bf16 row-major,
//      2 -> bf16 transposed [b][feature][seq] (for attention V)
// ---------------------------------------------------------------------------
template <int ACT, int OB>
__global__ __launch_bounds__(128) void wmma_gemm(const bf16_t* __restrict__ A,
                                                 const bf16_t* __restrict__ Wt,
                                                 const float* __restrict__ bias,
                                                 const float* __restrict__ res,
                                                 void* __restrict__ Cp, int M,
                                                 int N, int K) {
  const int lane = threadIdx.x & 31;
  const int wv = threadIdx.x >> 5;
  const int hf = lane >> 4, lr = lane & 15;
  const int tilesN = N >> 4;
  const int tile = blockIdx.x * 4 + wv;
  const int tm = tile / tilesN, tn = tile % tilesN;
  if (tm * 16 >= M) return;

  v8f acc = {};
  const bf16_t* Arow = A + (long)(tm * 16 + lr) * K;
  const bf16_t* Wrow = Wt + (long)(tn * 16 + lr) * K;
  for (int k0 = 0; k0 < K; k0 += 32) {
    V16 af, bw;
    af.v    = *(const v16bf*)(Arow + k0 + hf * 16);
    bw.v    = *(const v16bf*)(Wrow + k0 + hf * 16);
    if (k0 + 32 < K) __builtin_prefetch(Arow + k0 + 256, 0, 0);
    acc = wmma_bf(af.v, bw.v, acc);
  }

#pragma unroll
  for (int r = 0; r < 8; ++r) {
    const int m = tm * 16 + r + hf * 8;
    const int n = tn * 16 + lr;
    float v = acc[r];
    if (bias) v += bias[n];
    if (ACT == 1) {  // gelu (tanh approx, jax.nn.gelu default)
      const float c = 0.7978845608028654f * (v + 0.044715f * v * v * v);
      v = 0.5f * v * (1.0f + tanhf(c));
    } else if (ACT == 2) {
      v = 1.0f / (1.0f + __expf(-v));
    }
    if (res) v += res[(long)m * N + n];
    if (OB == 0) {
      ((float*)Cp)[(long)m * N + n] = v;
    } else if (OB == 1) {
      ((bf16_t*)Cp)[(long)m * N + n] = (__bf16)v;
    } else {  // transposed: [b][feature n][seq]
      const int bb = m >> 11;        // m / S_
      const int ss = m & (S_ - 1);
      ((bf16_t*)Cp)[(((long)bb * N) + n) * S_ + ss] = (__bf16)v;
    }
  }
}

// ---------------------------------------------------------------------------
// phi map: pq[row, 0:512] = q, pq[row, 512:1024] = q^2  (bf16)
// ---------------------------------------------------------------------------
__global__ __launch_bounds__(256) void phi_map(const float* __restrict__ q,
                                               bf16_t* __restrict__ pq,
                                               int total) {
  const int i = blockIdx.x * 256 + threadIdx.x;
  if (i >= total) return;
  const int row = i >> 9, c = i & 511;
  const float v = q[i];
  pq[(long)row * P_ + c]       = (__bf16)v;
  pq[(long)row * P_ + 512 + c] = (__bf16)(v * v);
}

// ---------------------------------------------------------------------------
// Omega scan. Grid: B*(D/16)=64 blocks, 256 threads (8 waves).
// M,St in VGPRs; double-buffered pw (TDM) and pqt (async) prefetched one
// step ahead so DMA overlaps compute.
// ---------------------------------------------------------------------------
__global__ __launch_bounds__(256) void omega_scan(const bf16_t* __restrict__ pk,
                                                  const bf16_t* __restrict__ pq,
                                                  const float* __restrict__ v,
                                                  bf16_t* __restrict__ ys) {
  __shared__ __bf16 Mbft[16 * P_];         // 32 KB (bf16 shadow of M, [d][p])
  __shared__ __bf16 pwb[2 * CW_ * P_];     // 64 KB (double-buffered window)
  __shared__ __bf16 pqtb[2 * P_];          // 4 KB
  __shared__ float  epart[8 * 16 * 16];    // 8 KB
  __shared__ __bf16 esum_t[16 * 16];       // [d][w]
  __shared__ float  vw[CW_ * 16];
  __shared__ float  ypart[16 * 16];

  const int tid = threadIdx.x;
  const int lane = tid & 31, wv = tid >> 5;
  const int hf = lane >> 4, lr = lane & 15;
  const int b = blockIdx.x >> 5;  // D/16 = 32 slices per batch
  const int dblk = blockIdx.x & 31;
  const int d0 = dblk * 16;

  v8f Mreg[8], Sreg[8];
  {
    v8f z = {};
#pragma unroll
    for (int i = 0; i < 8; ++i) { Mreg[i] = z; Sreg[i] = z; }
  }
  for (int i = tid; i < 16 * P_; i += 256) Mbft[i] = (__bf16)0.f;
  __syncthreads();

  const uint32_t pwLds  = (uint32_t)(uintptr_t)pwb;   // LDS addr = flat[31:0]
  const uint32_t pqtLds = (uint32_t)(uintptr_t)pqtb;

  for (int t = 0; t < S_; ++t) {
    const int s0 = t - (CW_ - 1);
    const int tn1 = t + 1;
    __bf16* pwc = pwb + (t & 1) * (CW_ * P_);
    __bf16* pqc = pqtb + (t & 1) * P_;

#if HAS_TDM_
    // ---- prefetch NEXT step (DMA overlaps this step's compute) -----------
    if (tn1 < S_) {
      if (tid < 128) {  // pqt(t+1): 2 KB = 128 lanes x 16B async
        const uint32_t gof =
            (uint32_t)((((long)b * S_ + tn1) * P_) * 2 + tid * 16);
        const uint32_t lds =
            pqtLds + (uint32_t)((tn1 & 1) * (P_ * 2) + tid * 16);
        asm volatile("global_load_async_to_lds_b128 %0, %1, %2"
                     :: "v"(lds), "v"(gof), "s"(pq) : "memory");
      }
      if (tn1 >= CW_ - 1 && wv == 0) {  // pw(t+1) via TDM
        const unsigned long long ga = (unsigned long long)(uintptr_t)(
            pk + ((long)b * S_ + (tn1 - (CW_ - 1))) * P_);
        v4u g0;
        g0[0] = 1u;                          // count=1 (valid user descriptor)
        g0[1] = pwLds + (uint32_t)((tn1 & 1) * (CW_ * P_ * 2));
        g0[2] = (unsigned int)ga;            // global_addr[31:0]
        g0[3] = (unsigned int)((ga >> 32) & 0x1FFFFFFu) | (2u << 30);  // type=2
        v8i g1 = {};
        g1[0] = (int)(1u << 16);             // data_size=1 (2 bytes/elem)
        g1[1] = (int)(1024u << 16);          // tensor_dim0 = 1024
        g1[2] = (int)(2048u << 16);          // tensor_dim1 = 2048
        g1[3] = (int)(1024u << 16);          // tile_dim0 = 1024
        g1[4] = 16;                          // tile_dim1 = 16
        g1[5] = 1024;                        // tensor_dim0_stride = 1024
        v4i g2 = {}, g3 = {};
#if defined(__clang_major__) && __clang_major__ >= 23
        v8i g4 = {};
        __builtin_amdgcn_tensor_load_to_lds(g0, g1, g2, g3, g4, 0);
#else
        __builtin_amdgcn_tensor_load_to_lds(g0, g1, g2, g3, 0);
#endif
      }
    }
    // ---- make CURRENT step's operands resident ---------------------------
    if (t == 0) {
      for (int i = tid; i < P_; i += 256)
        pqc[i] = pq[((long)b * S_) * P_ + i];
    } else {
      // async loads retire in order: <=1 leaves only the t+1 prefetch
      if (tn1 < S_) asm volatile("s_wait_asynccnt 0x1" ::: "memory");
      else          asm volatile("s_wait_asynccnt 0x0" ::: "memory");
    }
    if (s0 >= 0) {
      if (wv == 0) {
        if (tn1 < S_) __builtin_amdgcn_s_wait_tensorcnt(1);
        else          __builtin_amdgcn_s_wait_tensorcnt(0);
      }
    } else {
      for (int i = tid; i < CW_ * P_; i += 256) {
        const int w = i >> 10, p = i & 1023;
        const int s = s0 + w;
        pwc[w * P_ + p] = (s >= 0) ? pk[((long)b * S_ + s) * P_ + p]
                                   : (__bf16)0.f;
      }
    }
#else
    // ---- fallback: synchronous per-lane async staging of current step ----
    if (s0 >= 0) {
      const uint32_t gbase = (uint32_t)(((long)b * S_ + s0) * P_ * 2);
#pragma unroll
      for (int it = 0; it < 8; ++it) {
        const int i = tid + it * 256;
        const int w = i >> 7, off = i & 127;
        const uint32_t lds = pwLds + (uint32_t)((t & 1) * (CW_ * P_ * 2) +
                                                w * (P_ * 2) + off * 16);
        const uint32_t gof = gbase + (uint32_t)(w * (P_ * 2) + off * 16);
        asm volatile("global_load_async_to_lds_b128 %0, %1, %2"
                     :: "v"(lds), "v"(gof), "s"(pk) : "memory");
      }
      if (tid < 128) {
        const uint32_t gof =
            (uint32_t)((((long)b * S_ + t) * P_) * 2 + tid * 16);
        const uint32_t lds = pqtLds + (uint32_t)((t & 1) * (P_ * 2) + tid * 16);
        asm volatile("global_load_async_to_lds_b128 %0, %1, %2"
                     :: "v"(lds), "v"(gof), "s"(pq) : "memory");
      }
      asm volatile("s_wait_asynccnt 0x0" ::: "memory");
    } else {
      for (int i = tid; i < CW_ * P_; i += 256) {
        const int w = i >> 10, p = i & 1023;
        const int s = s0 + w;
        pwc[w * P_ + p] = (s >= 0) ? pk[((long)b * S_ + s) * P_ + p]
                                   : (__bf16)0.f;
      }
      for (int i = tid; i < P_; i += 256)
        pqc[i] = pq[((long)b * S_ + t) * P_ + i];
    }
#endif
    {
      const int w = tid >> 4, d = tid & 15;
      const int s = s0 + w;
      vw[w * 16 + d] = (s >= 0) ? v[((long)b * S_ + s) * D_ + d0 + d] : 0.f;
    }
    __syncthreads();

    // ---- err partials: wave wv covers K in [wv*128, +128), 4 wmma --------
    {
      v8f acc = {};
      const __bf16* pwr = pwc + lr * P_;
      const __bf16* mbr = Mbft + lr * P_;
      for (int kc = 0; kc < 4; ++kc) {
        const int k0 = wv * 128 + kc * 32;
        V16 af, bm;
        af.v    = *(const v16bf*)(pwr + k0 + hf * 16);
        bm.v    = *(const v16bf*)(mbr + k0 + hf * 16);
        acc = wmma_bf(af.v, bm.v, acc);
      }
#pragma unroll
      for (int r = 0; r < 8; ++r)
        epart[(wv * 16 + r + hf * 8) * 16 + lr] = acc[r];
    }
    __syncthreads();

    // ---- reduce partials, subtract vw; store transposed bf16 -------------
    {
      const int w = tid >> 4, d = tid & 15;
      float e = -vw[w * 16 + d];
#pragma unroll
      for (int j = 0; j < 8; ++j) e += epart[(j * 16 + w) * 16 + d];
      esum_t[d * 16 + w] = (__bf16)e;
    }
    __syncthreads();

    // ---- grad (pw^T @ err, K=16 padded); M/St register update ------------
    {
      V16 be;
      be.h[0] = *(const v8bf*)&esum_t[lr * 16 + hf * 8];
#pragma unroll
      for (int j = 8; j < 16; ++j) be.e[j] = (__bf16)0.f;
#pragma unroll
      for (int i = 0; i < 8; ++i) {
        const int p0 = (wv * 8 + i) * 16;
        V16 af;
#pragma unroll
        for (int j = 0; j < 8; ++j)
          af.e[j] = pwc[(hf * 8 + j) * P_ + p0 + lr];
#pragma unroll
        for (int j = 8; j < 16; ++j) af.e[j] = (__bf16)0.f;
        v8f zf = {};
        const v8f g = wmma_bf(af.v, be.v, zf);
        Sreg[i] = 0.9f * Sreg[i] + g * (1.0f / 16.0f);
        Mreg[i] = 0.99f * Mreg[i] - 0.1f * Sreg[i];
        V8 mp;
#pragma unroll
        for (int r = 0; r < 8; ++r) mp.e[r] = (__bf16)Mreg[i][r];
        *(v8bf*)(Mbft + lr * P_ + p0 + hf * 8) = mp.v;  // packed 16B store
      }
    }
    __syncthreads();

    // ---- y_t = pq_t @ M (from bf16 shadow; P split over 16 groups) -------
    {
      const int d = tid & 15, grp = tid >> 4;
      const __bf16* mrow = Mbft + d * P_;
      float ps = 0.f;
      for (int p = grp * 64; p < grp * 64 + 64; ++p)
        ps += (float)pqc[p] * (float)mrow[p];
      ypart[grp * 16 + d] = ps;
    }
    __syncthreads();
    if (tid < 16) {
      float s = 0.f;
#pragma unroll
      for (int g2 = 0; g2 < 16; ++g2) s += ypart[g2 * 16 + tid];
      ys[((long)b * S_ + t) * D_ + d0 + tid] = (__bf16)s;
    }
    __syncthreads();
  }
}

// ---------------------------------------------------------------------------
// Sliding-window causal attention (flash-style). One wave per q-tile.
// Q/K row-major bf16; V transposed [b][feature][seq] -> 16B vector B-frags.
// ---------------------------------------------------------------------------
__global__ __launch_bounds__(128) void swa_kernel(const bf16_t* __restrict__ qa,
                                                  const bf16_t* __restrict__ ka,
                                                  const bf16_t* __restrict__ vat,
                                                  bf16_t* __restrict__ op) {
  __shared__ __bf16 pbuf[4][16][16];
  const int lane = threadIdx.x & 31, wv = threadIdx.x >> 5;
  const int hf = lane >> 4, lr = lane & 15;
  const int tile = blockIdx.x * 4 + wv;  // B*H*(S/16) = 1024 tiles
  const int qt = tile & 127;
  const int h  = (tile >> 7) & 3;
  const int b  = tile >> 9;
  const long base = (long)b * S_ * D_ + h * 128;

  V16 qf[4];
  {
    const bf16_t* Qr = qa + base + (long)(qt * 16 + lr) * D_;
#pragma unroll
    for (int kc = 0; kc < 4; ++kc) {
      qf[kc].v = *(const v16bf*)(Qr + kc * 32 + hf * 16);
    }
  }

  v8f o_[8];
  float l_[8], mx_[8];
  {
    v8f z = {};
#pragma unroll
    for (int db = 0; db < 8; ++db) o_[db] = z;
#pragma unroll
    for (int r = 0; r < 8; ++r) { l_[r] = 0.f; mx_[r] = -1e30f; }
  }

  int kt_lo = qt * 16 - (WIN_ - 1);
  kt_lo = (kt_lo < 0) ? 0 : (kt_lo >> 4);
  for (int kt = kt_lo; kt <= qt; ++kt) {
    v8f s = {};
    const bf16_t* Kr = ka + base + (long)(kt * 16 + lr) * D_;
#pragma unroll
    for (int kc = 0; kc < 4; ++kc) {
      V16 bk;
      bk.v = *(const v16bf*)(Kr + kc * 32 + hf * 16);
      s = wmma_bf(qf[kc].v, bk.v, s);
    }

    float pnew[8], scl[8];
#pragma unroll
    for (int r = 0; r < 8; ++r) {
      const int gi = qt * 16 + r + hf * 8;
      const int gj = kt * 16 + lr;
      const bool ok = (gj <= gi) && (gi - gj < WIN_);
      float sv = ok ? s[r] * 0.088388347648318f : -1e30f;  // dh^-0.5
      float rm = sv;
#pragma unroll
      for (int msk = 1; msk < 16; msk <<= 1)
        rm = fmaxf(rm, __shfl_xor(rm, msk, 32));
      const float mnew = fmaxf(mx_[r], rm);
      scl[r] = __expf(mx_[r] - mnew);
      float p = ok ? __expf(sv - mnew) : 0.f;
      float rs = p;
#pragma unroll
      for (int msk = 1; msk < 16; msk <<= 1) rs += __shfl_xor(rs, msk, 32);
      l_[r] = l_[r] * scl[r] + rs;
      mx_[r] = mnew;
      pnew[r] = p;
    }
#pragma unroll
    for (int db = 0; db < 8; ++db)
#pragma unroll
      for (int r = 0; r < 8; ++r) o_[db][r] *= scl[r];

    // stage P (bf16) through LDS; per-wave, DS ops in order
#pragma unroll
    for (int r = 0; r < 8; ++r)
      pbuf[wv][r + hf * 8][lr] = (__bf16)pnew[r];
    asm volatile("s_wait_dscnt 0x0" ::: "memory");

    V16 pf;
    pf.h[0] = *(const v8bf*)&pbuf[wv][lr][hf * 8];
#pragma unroll
    for (int j = 8; j < 16; ++j) pf.e[j] = (__bf16)0.f;

    // O += P @ V : V^T row = feature, contiguous over keys -> one 16B load
#pragma unroll
    for (int db = 0; db < 8; ++db) {
      V16 vf;
      vf.h[0] = *(const v8bf*)(vat +
            (((long)b * D_) + h * 128 + db * 16 + lr) * S_ + kt * 16 + hf * 8);
#pragma unroll
      for (int j = 8; j < 16; ++j) vf.e[j] = (__bf16)0.f;
      o_[db] = wmma_bf(pf.v, vf.v, o_[db]);
    }
  }

#pragma unroll
  for (int db = 0; db < 8; ++db)
#pragma unroll
    for (int r = 0; r < 8; ++r) {
      const int m = qt * 16 + r + hf * 8;
      op[base + (long)m * D_ + db * 16 + lr] = (__bf16)(o_[db][r] / l_[r]);
    }
}

// ---------------------------------------------------------------------------
// Elementwise helpers
// ---------------------------------------------------------------------------
__global__ __launch_bounds__(256) void concat2(const float* __restrict__ a,
                                               const float* __restrict__ b,
                                               bf16_t* __restrict__ cc,
                                               int total) {
  const int i = blockIdx.x * 256 + threadIdx.x;
  if (i >= total) return;
  const int row = i >> 9, c = i & 511;
  cc[(long)row * 1024 + c]       = (__bf16)a[i];
  cc[(long)row * 1024 + 512 + c] = (__bf16)b[i];
}

__global__ __launch_bounds__(256) void combine_gate(const float* __restrict__ x,
                                                    const float* __restrict__ gt,
                                                    const float* __restrict__ mo,
                                                    const float* __restrict__ ao,
                                                    float* __restrict__ h,
                                                    int n) {
  const int i = blockIdx.x * 256 + threadIdx.x;
  if (i >= n) return;
  const float g = gt[i];
  h[i] = x[i] + g * mo[i] + (1.0f - g) * ao[i];
}

// ---------------------------------------------------------------------------
// Host launcher
// ---------------------------------------------------------------------------
extern "C" void kernel_launch(void* const* d_in, const int* in_sizes, int n_in,
                              void* d_out, int out_size, void* d_ws,
                              size_t ws_size, hipStream_t stream) {
  const float* x   = (const float*)d_in[0];
  const float* g1  = (const float*)d_in[1];
  const float* b1  = (const float*)d_in[2];
  const float* g2  = (const float*)d_in[3];
  const float* b2  = (const float*)d_in[4];
  const float* g3  = (const float*)d_in[5];
  const float* b3  = (const float*)d_in[6];
  const float* Wmq = (const float*)d_in[7];
  const float* Wmk = (const float*)d_in[8];
  const float* Wmv = (const float*)d_in[9];
  const float* Wmo = (const float*)d_in[10];
  const float* Waq = (const float*)d_in[11];
  const float* Wak = (const float*)d_in[12];
  const float* Wav = (const float*)d_in[13];
  const float* Wao = (const float*)d_in[14];
  const float* Wg  = (const float*)d_in[15];
  const float* bg  = (const float*)d_in[16];
  const float* W1  = (const float*)d_in[17];
  const float* bf1 = (const float*)d_in[18];
  const float* W2  = (const float*)d_in[19];
  const float* bf2 = (const float*)d_in[20];
  float* out = (float*)d_out;

  const long SZ = (long)B_ * S_ * D_;  // 2M elements
  float* f = (float*)d_ws;
  float* qb   = f + 0 * SZ;
  float* kb   = f + 1 * SZ;
  float* vb   = f + 2 * SZ;
  float* memo = f + 3 * SZ;
  float* aout = f + 4 * SZ;
  float* gate = f + 5 * SZ;
  float* hbuf = f + 6 * SZ;
  bf16_t* bp    = (bf16_t*)(f + 7 * SZ);
  bf16_t* xn16  = bp;            bp += SZ;
  bf16_t* ys16  = bp;            bp += SZ;
  bf16_t* qa16  = bp;            bp += SZ;
  bf16_t* ka16  = bp;            bp += SZ;
  bf16_t* vat16 = bp;            bp += SZ;   // V transposed [b][feat][seq]
  bf16_t* ap16  = bp;            bp += SZ;
  bf16_t* cc16  = bp;            bp += 2 * SZ;
  bf16_t* t116  = bp;            bp += 4 * SZ;
  bf16_t* pqb   = bp;            bp += 2 * SZ;
  bf16_t* pkb   = bp;            bp += 2 * SZ;
  bf16_t* wmq_t = bp;            bp += 512L * 512;
  bf16_t* wmk_t = bp;            bp += 512L * 512;
  bf16_t* wmv_t = bp;            bp += 512L * 512;
  bf16_t* wmo_t = bp;            bp += 512L * 512;
  bf16_t* waq_t = bp;            bp += 512L * 512;
  bf16_t* wak_t = bp;            bp += 512L * 512;
  bf16_t* wav_t = bp;            bp += 512L * 512;
  bf16_t* wao_t = bp;            bp += 512L * 512;
  bf16_t* wg_t  = bp;            bp += 1024L * 512;
  bf16_t* w1_t  = bp;            bp += 512L * 2048;
  bf16_t* w2_t  = bp;            bp += 2048L * 512;

  const int ROWS = B_ * S_;  // 4096
  const int EW = (int)SZ;    // 2M
  const int ewg = (EW + 255) / 256;
  auto ggrid = [](int N) { return (4096 / 16) * (N / 16) / 4; };
  dim3 b256(256), b128(128);

  // --- weight transpose/convert ---
  tw_kernel<<<1024, b256, 0, stream>>>(Wmq, wmq_t, 512, 512);
  tw_kernel<<<1024, b256, 0, stream>>>(Wmk, wmk_t, 512, 512);
  tw_kernel<<<1024, b256, 0, stream>>>(Wmv, wmv_t, 512, 512);
  tw_kernel<<<1024, b256, 0, stream>>>(Wmo, wmo_t, 512, 512);
  tw_kernel<<<1024, b256, 0, stream>>>(Waq, waq_t, 512, 512);
  tw_kernel<<<1024, b256, 0, stream>>>(Wak, wak_t, 512, 512);
  tw_kernel<<<1024, b256, 0, stream>>>(Wav, wav_t, 512, 512);
  tw_kernel<<<1024, b256, 0, stream>>>(Wao, wao_t, 512, 512);
  tw_kernel<<<2048, b256, 0, stream>>>(Wg, wg_t, 1024, 512);
  tw_kernel<<<4096, b256, 0, stream>>>(W1, w1_t, 512, 2048);
  tw_kernel<<<4096, b256, 0, stream>>>(W2, w2_t, 2048, 512);

  // --- branch 1: ln1 -> q,k,v -> phi -> omega scan -> Wmo ---
  ln512<<<ROWS, b256, 0, stream>>>(x, g1, b1, xn16);
  wmma_gemm<0, 0><<<ggrid(512), b128, 0, stream>>>(xn16, wmq_t, nullptr, nullptr, qb, ROWS, 512, 512);
  wmma_gemm<0, 0><<<ggrid(512), b128, 0, stream>>>(xn16, wmk_t, nullptr, nullptr, kb, ROWS, 512, 512);
  wmma_gemm<0, 0><<<ggrid(512), b128, 0, stream>>>(xn16, wmv_t, nullptr, nullptr, vb, ROWS, 512, 512);
  phi_map<<<ewg, b256, 0, stream>>>(qb, pqb, EW);
  phi_map<<<ewg, b256, 0, stream>>>(kb, pkb, EW);
  omega_scan<<<B_ * (D_ / 16), b256, 0, stream>>>(pkb, pqb, vb, ys16);
  wmma_gemm<0, 0><<<ggrid(512), b128, 0, stream>>>(ys16, wmo_t, nullptr, nullptr, memo, ROWS, 512, 512);

  // --- branch 2: ln2 -> q,k,v -> sliding-window attention -> Wao ---
  ln512<<<ROWS, b256, 0, stream>>>(x, g2, b2, xn16);
  wmma_gemm<0, 1><<<ggrid(512), b128, 0, stream>>>(xn16, waq_t, nullptr, nullptr, qa16, ROWS, 512, 512);
  wmma_gemm<0, 1><<<ggrid(512), b128, 0, stream>>>(xn16, wak_t, nullptr, nullptr, ka16, ROWS, 512, 512);
  wmma_gemm<0, 2><<<ggrid(512), b128, 0, stream>>>(xn16, wav_t, nullptr, nullptr, vat16, ROWS, 512, 512);
  swa_kernel<<<(B_ * 4 * (S_ / 16)) / 4, b128, 0, stream>>>(qa16, ka16, vat16, ap16);
  wmma_gemm<0, 0><<<ggrid(512), b128, 0, stream>>>(ap16, wao_t, nullptr, nullptr, aout, ROWS, 512, 512);

  // --- gate + residual ---
  concat2<<<ewg, b256, 0, stream>>>(memo, aout, cc16, EW);
  wmma_gemm<2, 0><<<ggrid(512), b128, 0, stream>>>(cc16, wg_t, bg, nullptr, gate, ROWS, 512, 1024);
  combine_gate<<<ewg, b256, 0, stream>>>(x, gate, memo, aout, hbuf, EW);

  // --- ln3 -> FFN -> residual ---
  ln512<<<ROWS, b256, 0, stream>>>(hbuf, g3, b3, xn16);
  wmma_gemm<1, 1><<<ggrid(DF_), b128, 0, stream>>>(xn16, w1_t, bf1, nullptr, t116, ROWS, DF_, 512);
  wmma_gemm<0, 0><<<ggrid(512), b128, 0, stream>>>(t116, w2_t, bf2, hbuf, out, ROWS, 512, DF_);
}
